// SalAttention_backup_83193516523630
// MI455X (gfx1250) — matrix operand order repaired and checked by reading, not compiled, not run on hardware
//
#include <hip/hip_runtime.h>

// ---------------------------------------------------------------------------
// SalAttention (RBF-kernel attention w/ RPE bias) for gfx1250 (MI455X).
// All GEMMs use v_wmma_f32_16x16x32_bf16 (fp32 accumulate).
// Pipeline:
//   1. pack x -> bf16, pack Wq/Wk/Wv/Wp -> bf16 transposed [N][K]
//   2. bias  : biasH[h][i][j] = gelu(rpe_table[rpe_index[i][j]][h])
//   3. gemm  : Q = x Wq, K = x Wk, V = x Wv   (WMMA; V stored head-transposed)
//   4. norms : q2/k2 row sums of squares
//   5. attn  : S = q k^T (WMMA) -> rbf logits - bias -> softmax -> O = P V (WMMA)
//   6. gemm  : out = O Wp + bp  (WMMA, fp32 store)
// Workspace use: ~102 MiB.
// ---------------------------------------------------------------------------

typedef __attribute__((ext_vector_type(16))) __bf16 v16bf;
typedef __attribute__((ext_vector_type(8)))  __bf16 v8bf;
typedef __attribute__((ext_vector_type(8)))  float  v8f;

#define BATCH 64
#define NP    196          // patches
#define CH    768          // embed dim
#define NH    12           // heads
#define HD    64           // head dim
#define MROWS (BATCH*NP)   // 12544
#define NKP   224          // padded K for P@V (7 x 32)
#define RBF   (-0.0625f)   // -0.5 * 64^-0.5

__device__ __forceinline__ v16bf cat8(v8bf lo, v8bf hi) {
  return __builtin_shufflevector(lo, hi, 0,1,2,3,4,5,6,7,8,9,10,11,12,13,14,15);
}
__device__ __forceinline__ v8f wmma_bf16(v16bf a, v16bf b, v8f c) {
  return __builtin_amdgcn_wmma_f32_16x16x32_bf16(false, a, false, b, (short)0, c,
                                                 false, false);
}

// ---- pack kernels ---------------------------------------------------------
__global__ void pack_x_kernel(const float* __restrict__ x, __bf16* __restrict__ xb,
                              int n) {
  int i = blockIdx.x * blockDim.x + threadIdx.x;
  if (i < n) xb[i] = (__bf16)x[i];
}

// wt holds 4 transposed matrices back to back: [w][n][k] = W[k][n]
__global__ void pack_wt_kernel(const float* __restrict__ Wq, const float* __restrict__ Wk,
                               const float* __restrict__ Wv, const float* __restrict__ Wp,
                               __bf16* __restrict__ wt) {
  int i = blockIdx.x * blockDim.x + threadIdx.x;
  const int per = CH * CH;
  if (i >= 4 * per) return;
  int w = i / per, rem = i % per;
  int n = rem / CH, k = rem % CH;
  const float* src = (w == 0) ? Wq : (w == 1) ? Wk : (w == 2) ? Wv : Wp;
  wt[i] = (__bf16)src[(size_t)k * CH + n];
}

// biasH[h*NP*NP + i*NP + j] = gelu_exact(rpe_table[rpe_index[i*NP+j]*NH + h])
__global__ void bias_gelu_kernel(const float* __restrict__ table,
                                 const int* __restrict__ idx,
                                 float* __restrict__ biasH) {
  int t = blockIdx.x * blockDim.x + threadIdx.x;
  if (t >= NH * NP * NP) return;
  int h = t / (NP * NP);
  int ij = t % (NP * NP);
  float v = table[(size_t)idx[ij] * NH + h];
  biasH[t] = 0.5f * v * (1.0f + erff(v * 0.70710678118654752f));
}

__global__ void zero_vtpad_kernel(__bf16* __restrict__ Vt) {
  int t = blockIdx.x * blockDim.x + threadIdx.x;
  const int rows = BATCH * NH * HD;           // 49152
  const int pad  = NKP - NP;                  // 28
  if (t >= rows * pad) return;
  int row = t / pad, c = NP + t % pad;
  Vt[(size_t)row * NKP + c] = (__bf16)0.0f;
}

// ---- unified WMMA GEMM: C[M,768] = A[M,768] * Bt[768,768]^T ---------------
// block = 256 thr = 8 waves; each wave owns one 16x16 tile.
// grid.x = 768/128 = 6 col-groups, grid.y = M/16 row tiles.
// mode 0: store bf16 Q  [b][h][n][d]
// mode 1: store bf16 K  [b][h][n][d]
// mode 2: store bf16 Vt [b][h][d][n] with row stride NKP
// mode 3: store f32 out[m][c] + bias[c]
__global__ __launch_bounds__(256)
void gemm_bf16_kernel(const __bf16* __restrict__ A, const __bf16* __restrict__ Bt,
                      void* __restrict__ out, const float* __restrict__ bias,
                      int mode) {
  const int wave = threadIdx.x >> 5;
  const int lane = threadIdx.x & 31;
  const int ln = lane & 15, hi = lane >> 4, khalf = hi * 8;
  const int mbase = blockIdx.y * 16;
  const int nbase = (blockIdx.x * 8 + wave) * 16;

  const __bf16* arow = A + (size_t)(mbase + ln) * CH;
  const __bf16* brow = Bt + (size_t)(nbase + ln) * CH + khalf;

  v8f c = {};
  for (int k0 = 0; k0 < CH; k0 += 32) {
    __builtin_prefetch(arow + k0 + 128, 0, 3);   // global_prefetch_b8
    __builtin_prefetch(brow + k0 + 128, 0, 3);
    v8bf alo = *(const v8bf*)(arow + k0 + khalf);
    v8bf ahi = *(const v8bf*)(arow + k0 + 16 + khalf);
    v16bf a = cat8(alo, ahi);
    v16bf b = cat8(*(const v8bf*)(brow + k0), *(const v8bf*)(brow + k0 + 16));
    c = wmma_bf16(a, b, c);
  }

  const int col = nbase + ln;
  const int h = col >> 6, d = col & 63;
#pragma unroll
  for (int r = 0; r < 8; ++r) {
    int m = mbase + r + hi * 8;
    float v = c[r];
    if (mode == 3) {
      ((float*)out)[(size_t)m * CH + col] = v + bias[col];
    } else {
      int bb = m / NP, nn = m % NP;
      __bf16* dst = (__bf16*)out;
      if (mode == 2)
        dst[((size_t)(bb * NH + h) * HD + d) * NKP + nn] = (__bf16)v;
      else
        dst[((size_t)(bb * NH + h) * NP + nn) * HD + d] = (__bf16)v;
    }
  }
}

// ---- per-head-row sums of squares ----------------------------------------
__global__ void rownorm_kernel(const __bf16* __restrict__ Qb,
                               const __bf16* __restrict__ Kb,
                               float* __restrict__ q2, float* __restrict__ k2) {
  const int rows = BATCH * NH * NP;   // 150528
  int gw = (blockIdx.x * blockDim.x + threadIdx.x) >> 5;
  int lane = threadIdx.x & 31;
  if (gw >= rows) return;
  const __bf16* qp = Qb + (size_t)gw * HD;
  const __bf16* kp = Kb + (size_t)gw * HD;
  float sq = 0.f, sk = 0.f;
#pragma unroll
  for (int d = lane; d < HD; d += 32) {
    float qv = (float)qp[d]; sq += qv * qv;
    float kv = (float)kp[d]; sk += kv * kv;
  }
#pragma unroll
  for (int off = 16; off; off >>= 1) {
    sq += __shfl_xor(sq, off);
    sk += __shfl_xor(sk, off);
  }
  if (lane == 0) { q2[gw] = sq; k2[gw] = sk; }
}

// ---- attention core: one block per (row-tile, head, batch) ----------------
__global__ __launch_bounds__(256)
void attn_kernel(const __bf16* __restrict__ Qb, const __bf16* __restrict__ Kb,
                 const __bf16* __restrict__ Vt, const float* __restrict__ q2,
                 const float* __restrict__ k2, const float* __restrict__ biasH,
                 __bf16* __restrict__ Ob) {
  __shared__ __align__(16) float  S[16][208];
  __shared__ __align__(16) __bf16 P[16][NKP];

  const int it = blockIdx.x;      // 0..12 row tiles
  const int h  = blockIdx.y;      // head
  const int b  = blockIdx.z;      // batch
  const int wave = threadIdx.x >> 5;
  const int lane = threadIdx.x & 31;
  const int ln = lane & 15, hi = lane >> 4, khalf = hi * 8;
  const int ibase = it * 16;
  const size_t head = (size_t)(b * NH + h);
  const size_t headQ = head * NP * HD;

  // -- S = q k^T (13 column tiles over 8 waves, K=64 => 2 WMMAs each)
  for (int jt = wave; jt < 13; jt += 8) {
    const int irow = ibase + ln;
    const int jrow = jt * 16 + ln;
    const __bf16* qp = Qb + headQ + (size_t)irow * HD + khalf;
    const __bf16* kp = Kb + headQ + (size_t)jrow * HD + khalf;
    v8f c = {};
#pragma unroll
    for (int k0 = 0; k0 < HD; k0 += 32) {
      v16bf a = {};
      if (irow < NP) a = cat8(*(const v8bf*)(qp + k0), *(const v8bf*)(qp + k0 + 16));
      v16bf bf = {};
      if (jrow < NP) bf = cat8(*(const v8bf*)(kp + k0), *(const v8bf*)(kp + k0 + 16));
      c = wmma_bf16(a, bf, c);
    }
#pragma unroll
    for (int r = 0; r < 8; ++r) S[r + hi * 8][jt * 16 + ln] = c[r];
  }
  __syncthreads();

  // -- rbf logits + bias + softmax; wave w owns rows w and w+8
  const float* bh = biasH + (size_t)h * NP * NP;
#pragma unroll
  for (int rr = 0; rr < 2; ++rr) {
    const int il = wave + rr * 8;
    const int i = ibase + il;
    if (i < NP) {
      const float q2i = q2[head * NP + i];
      float mmax = -3.0e38f;
      for (int j = lane; j < 208; j += 32) {
        float lg;
        if (j < NP) {
          lg = (q2i + k2[head * NP + j] - 2.0f * S[il][j]) * RBF
               - bh[(size_t)i * NP + j];
        } else {
          lg = -3.0e38f;
        }
        S[il][j] = lg;
        mmax = fmaxf(mmax, lg);
      }
#pragma unroll
      for (int off = 16; off; off >>= 1) mmax = fmaxf(mmax, __shfl_xor(mmax, off));
      float sum = 0.f;
      for (int j = lane; j < 208; j += 32) {
        float e = (j < NP) ? __expf(S[il][j] - mmax) : 0.0f;
        S[il][j] = e;
        sum += e;
      }
#pragma unroll
      for (int off = 16; off; off >>= 1) sum += __shfl_xor(sum, off);
      const float inv = 1.0f / sum;
      for (int j = lane; j < NKP; j += 32)
        P[il][j] = (__bf16)((j < NP) ? S[il][j] * inv : 0.0f);
    } else {
      for (int j = lane; j < NKP; j += 32) P[il][j] = (__bf16)0.0f;
    }
  }
  __syncthreads();

  // -- O = P @ V : 4 column tiles (d) on waves 0..3, K = 224 => 7 WMMAs
  if (wave < 4) {
    const int d = wave * 16 + ln;
    const __bf16* vp = Vt + (head * HD + d) * NKP + khalf;
    v8f c = {};
#pragma unroll
    for (int k0 = 0; k0 < NKP; k0 += 32) {
      v16bf a = cat8(*(const v8bf*)(&P[ln][k0 + khalf]),
                     *(const v8bf*)(&P[ln][k0 + 16 + khalf]));
      v16bf bf = cat8(*(const v8bf*)(vp + k0), *(const v8bf*)(vp + k0 + 16));
      c = wmma_bf16(a, bf, c);
    }
#pragma unroll
    for (int r = 0; r < 8; ++r) {
      int i = ibase + r + hi * 8;
      if (i < NP)
        Ob[((size_t)b * NP + i) * CH + h * HD + d] = (__bf16)c[r];
    }
  }
}

// ---------------------------------------------------------------------------
extern "C" void kernel_launch(void* const* d_in, const int* in_sizes, int n_in,
                              void* d_out, int out_size, void* d_ws, size_t ws_size,
                              hipStream_t stream) {
  (void)in_sizes; (void)n_in; (void)out_size; (void)ws_size;
  const float* x         = (const float*)d_in[0];
  const float* Wq        = (const float*)d_in[1];
  const float* Wk        = (const float*)d_in[2];
  const float* Wv        = (const float*)d_in[3];
  const float* rpe_table = (const float*)d_in[4];
  const float* Wp        = (const float*)d_in[5];
  const float* bp        = (const float*)d_in[6];
  const int*   rpe_index = (const int*)d_in[7];

  char* w = (char*)d_ws;
  __bf16* xb  = (__bf16*)w;  w += (size_t)MROWS * CH * 2;       // 19.3 MB
  __bf16* wt  = (__bf16*)w;  w += (size_t)4 * CH * CH * 2;      // 4.7 MB (q,k,v,p)
  __bf16* Qb  = (__bf16*)w;  w += (size_t)MROWS * CH * 2;
  __bf16* Kb  = (__bf16*)w;  w += (size_t)MROWS * CH * 2;
  __bf16* Vt  = (__bf16*)w;  w += (size_t)BATCH * NH * HD * NKP * 2;  // 22 MB
  float*  q2  = (float*)w;   w += (size_t)BATCH * NH * NP * 4;
  float*  k2  = (float*)w;   w += (size_t)BATCH * NH * NP * 4;
  float*  biasH = (float*)w; w += (size_t)NH * NP * NP * 4;
  __bf16* Ob  = (__bf16*)w;  w += (size_t)MROWS * CH * 2;
  __bf16* wqt = wt;
  __bf16* wkt = wt + (size_t)CH * CH;
  __bf16* wvt = wt + (size_t)2 * CH * CH;
  __bf16* wpt = wt + (size_t)3 * CH * CH;

  // 1) packing
  {
    int n = MROWS * CH;
    pack_x_kernel<<<(n + 255) / 256, 256, 0, stream>>>(x, xb, n);
    int nw = 4 * CH * CH;
    pack_wt_kernel<<<(nw + 255) / 256, 256, 0, stream>>>(Wq, Wk, Wv, Wp, wt);
    int nb = NH * NP * NP;
    bias_gelu_kernel<<<(nb + 255) / 256, 256, 0, stream>>>(rpe_table, rpe_index, biasH);
    int nz = BATCH * NH * HD * (NKP - NP);
    zero_vtpad_kernel<<<(nz + 255) / 256, 256, 0, stream>>>(Vt);
  }

  // 2) Q/K/V projections (WMMA)
  dim3 ggrid(CH / 128, MROWS / 16);           // (6, 784)
  gemm_bf16_kernel<<<ggrid, 256, 0, stream>>>(xb, wqt, Qb, nullptr, 0);
  gemm_bf16_kernel<<<ggrid, 256, 0, stream>>>(xb, wkt, Kb, nullptr, 1);
  gemm_bf16_kernel<<<ggrid, 256, 0, stream>>>(xb, wvt, Vt, nullptr, 2);

  // 3) row norms
  {
    int rows = BATCH * NH * NP;               // 150528, 8 waves per block
    rownorm_kernel<<<rows / 8, 256, 0, stream>>>(Qb, Kb, q2, k2);
  }

  // 4) attention (WMMA twice per block)
  attn_kernel<<<dim3(13, NH, BATCH), 256, 0, stream>>>(Qb, Kb, Vt, q2, k2, biasH, Ob);

  // 5) output projection + bias (WMMA, fp32 store)
  gemm_bf16_kernel<<<ggrid, 256, 0, stream>>>(Ob, wpt, (void*)d_out, bp, 3);
}
